// BatchProgramCC_3058016715393
// MI455X (gfx1250) — matrix operand order repaired and checked by reading, not compiled, not run on hardware
//
#include <hip/hip_runtime.h>
#include <hip/hip_bf16.h>

typedef __attribute__((ext_vector_type(16))) _Float16 v16h;
typedef __attribute__((ext_vector_type(8)))  _Float16 v8h;
typedef __attribute__((ext_vector_type(8)))  float    v8f;

#define WMMA_F32_F16(A,B,C) __builtin_amdgcn_wmma_f32_16x16x32_f16(false,(A),false,(B),(short)0,(C),false,false)

// Build a 16x32 f16 A-fragment from a row-major f16 tile with row stride
// `stride` halves: per lane, elements 0..15 = k0..k0+15 (k0 = 16*half).
__device__ __forceinline__ v16h load_a_frag(const _Float16* row, int kc, int half) {
    const _Float16* p = row + kc * 32 + 16 * half;
    v8h lo = *(const v8h*)(p);
    v8h hi = *(const v8h*)(p + 8);
    v16h af;
    #pragma unroll
    for (int i = 0; i < 8; ++i) { af[i] = lo[i]; af[i + 8] = hi[i]; }
    return af;
}

// ---------------------------------------------------------------------------
// f32 -> f16 weight conversion
// ---------------------------------------------------------------------------
__global__ void cvtK(const float* __restrict__ src, _Float16* __restrict__ dst, int n) {
    int i = blockIdx.x * 256 + threadIdx.x;
    if (i < n) dst[i] = (_Float16)src[i];
}

// ---------------------------------------------------------------------------
// Tree encoder: one block per (program, statement).
// Gather 63 embeddings -> LDS (f32), heap subtree sums, convert to f16 tile,
// 64x128x128 WMMA GEMM vs W_c^T, add subtreeCount*b_c, max over nodes.
// ---------------------------------------------------------------------------
__global__ void treeK(const int* __restrict__ tok1, const int* __restrict__ tok2,
                      const float* __restrict__ emb, const _Float16* __restrict__ Wc_h,
                      const float* __restrict__ b_c, _Float16* __restrict__ e_h) {
    __shared__ float    SH[64 * 129];             // f32 sums, later reused for H
    __shared__ _Float16 S16[64 * 136];            // f16 A tile (row stride 136)
    const int bx   = blockIdx.x;
    const int prog = bx >> 12;
    const int stmt = bx & 4095;
    const int* tok = (prog ? tok2 : tok1) + (size_t)stmt * 63;
    const int tid  = threadIdx.x;

    // Gather embeddings (row 63 zero-padded)
    for (int idx = tid; idx < 64 * 128; idx += 256) {
        int nn = idx >> 7, c = idx & 127;
        float v = 0.0f;
        if (nn < 63) v = emb[(size_t)tok[nn] * 128 + c];
        SH[nn * 129 + c] = v;
    }
    __syncthreads();
    // Bottom-up subtree sums: children of heap node p are 2p+1, 2p+2
    for (int d = 4; d >= 0; --d) {
        int ps = (1 << d) - 1, np = 1 << d;
        for (int idx = tid; idx < np * 128; idx += 256) {
            int pi = ps + (idx >> 7), c = idx & 127;
            SH[pi * 129 + c] += SH[(2 * pi + 1) * 129 + c] + SH[(2 * pi + 2) * 129 + c];
        }
        __syncthreads();
    }
    // Convert to f16 tile for vector fragment loads
    for (int idx = tid; idx < 64 * 128; idx += 256) {
        int nn = idx >> 7, c = idx & 127;
        S16[nn * 136 + c] = (_Float16)SH[nn * 129 + c];
    }
    __syncthreads();

    // GEMM: H[64x128] = S[64x128] @ Wc^T[128x128], wave w = N-tile w
    const int wv = tid >> 5, ln = tid & 31, half = ln >> 4, mn = ln & 15;
    v8f acc[4] = {};
    for (int kc = 0; kc < 4; ++kc) {
        v16h bf = *(const v16h*)(Wc_h + (size_t)(wv * 16 + mn) * 128 + kc * 32 + 16 * half);
        #pragma unroll
        for (int mt = 0; mt < 4; ++mt) {
            v16h af = load_a_frag(S16 + (mt * 16 + mn) * 136, kc, half);
            acc[mt] = WMMA_F32_F16(af, bf, acc[mt]);
        }
    }
    __syncthreads();            // all S reads done -> reuse SH for H
    #pragma unroll
    for (int mt = 0; mt < 4; ++mt)
        #pragma unroll
        for (int r = 0; r < 8; ++r)
            SH[(mt * 16 + r + 8 * half) * 129 + wv * 16 + mn] = acc[mt][r];
    __syncthreads();

    // Max over nodes 0..62 with per-level bias count
    if (tid < 128) {
        const int o = tid;
        const float bco = b_c[o];
        float mx = -1e30f;
        for (int m = 0; m < 63; ++m) {
            int   lvl = 31 - __clz(m + 1);
            float cnt = (float)((1 << (6 - lvl)) - 1);
            mx = fmaxf(mx, SH[m * 129 + o] + cnt * bco);
        }
        e_h[((size_t)prog * 4096 + stmt) * 128 + o] = (_Float16)mx;
    }
}

// ---------------------------------------------------------------------------
// xp = e @ W_ih^T + b_ih, for (prog,dir) pairs.  grid (64 Mblk, 6 Nblk, 4 pd)
// ---------------------------------------------------------------------------
__global__ void xpK(const _Float16* __restrict__ e_h, const _Float16* __restrict__ Wih_h,
                    const float* __restrict__ b_ih_f, const float* __restrict__ b_ih_b,
                    float* __restrict__ xp) {
    const int pd = blockIdx.z, prog = pd >> 1, dir = pd & 1;
    const _Float16* eh = e_h + (size_t)prog * 4096 * 128;
    const _Float16* W  = Wih_h + (size_t)dir * 98304;
    const float* bih   = dir ? b_ih_b : b_ih_f;
    float* xpo         = xp + (size_t)pd * 4096 * 768;

    const int tid = threadIdx.x, wv = tid >> 5, ln = tid & 31, half = ln >> 4, mn = ln & 15;
    const int n0 = blockIdx.y * 128 + wv * 16;
    const int m0 = blockIdx.x * 64;

    v8f acc[4] = {};
    for (int kc = 0; kc < 4; ++kc) {
        v16h bf = *(const v16h*)(W + (size_t)(n0 + mn) * 128 + kc * 32 + 16 * half);
        #pragma unroll
        for (int mt = 0; mt < 4; ++mt) {
            v16h af = load_a_frag(eh + (size_t)(m0 + mt * 16 + mn) * 128, kc, half);
            acc[mt] = WMMA_F32_F16(af, bf, acc[mt]);
        }
    }
    #pragma unroll
    for (int mt = 0; mt < 4; ++mt)
        #pragma unroll
        for (int r = 0; r < 8; ++r) {
            int m = m0 + mt * 16 + r + 8 * half;
            int col = n0 + mn;
            xpo[(size_t)m * 768 + col] = acc[mt][r] + bih[col];
        }
}

// ---------------------------------------------------------------------------
// Persistent GRU scan: one block per (prog,dir), 512 threads = 16 waves.
// Per step: gh = h @ W_hh^T (WMMA, A from f16 h-mirror), f32 gate math,
// running time-max.  Dynamic LDS:
//   h f32 [32][260] | gh f16 [32][784] | runmax f32 [32][256] | h16 [32][264]
// ---------------------------------------------------------------------------
__global__ void gruK(const float* __restrict__ xp, const _Float16* __restrict__ Whh_h,
                     const float* __restrict__ b_hh_f, const float* __restrict__ b_hh_b,
                     float* __restrict__ gmax) {
    extern __shared__ char smem[];
    float*    hS  = (float*)smem;                               // [32][260]
    _Float16* gS  = (_Float16*)(smem + 33280);                  // [32][784]
    float*    rm  = (float*)(smem + 33280 + 50176);             // [32][256]
    _Float16* h16 = (_Float16*)(smem + 33280 + 50176 + 32768);  // [32][264]

    const int pd = blockIdx.x, dir = pd & 1;
    const _Float16* W  = Whh_h + (size_t)dir * 196608;
    const float* bhh   = dir ? b_hh_b : b_hh_f;
    const float* xpp   = xp + (size_t)pd * 4096 * 768;
    float* gout        = gmax + (size_t)pd * 32 * 256;

    const int tid = threadIdx.x, wv = tid >> 5, ln = tid & 31, half = ln >> 4, mn = ln & 15;
    const int n0 = wv * 48;     // 3 N-tiles of 16 per wave, 16 waves -> 768 cols

    for (int idx = tid; idx < 32 * 260; idx += 512) hS[idx] = 0.0f;
    for (int idx = tid; idx < 32 * 264; idx += 512) h16[idx] = (_Float16)0.0f;
    for (int idx = tid; idx < 32 * 256; idx += 512) rm[idx] = -1e30f;
    __syncthreads();

    for (int s = 0; s < 128; ++s) {
        const int t = dir ? (127 - s) : s;
        v8f acc[2][3] = {};
        for (int kc = 0; kc < 8; ++kc) {
            v16h af0 = load_a_frag(h16 + (0 * 16 + mn) * 264, kc, half);
            v16h af1 = load_a_frag(h16 + (1 * 16 + mn) * 264, kc, half);
            #pragma unroll
            for (int nt = 0; nt < 3; ++nt) {
                v16h bf = *(const v16h*)(W + (size_t)(n0 + nt * 16 + mn) * 256 + kc * 32 + 16 * half);
                acc[0][nt] = WMMA_F32_F16(af0, bf, acc[0][nt]);
                acc[1][nt] = WMMA_F32_F16(af1, bf, acc[1][nt]);
            }
        }
        #pragma unroll
        for (int mt = 0; mt < 2; ++mt)
            #pragma unroll
            for (int nt = 0; nt < 3; ++nt)
                #pragma unroll
                for (int r = 0; r < 8; ++r)
                    gS[(mt * 16 + r + 8 * half) * 784 + n0 + nt * 16 + mn] = (_Float16)acc[mt][nt][r];
        __syncthreads();
        // Gates (PyTorch order r,z,n): each thread owns 16 (b,j) elements
        for (int idx = tid; idx < 32 * 256; idx += 512) {
            const int b = idx >> 8, j = idx & 255;
            const float* xrow = xpp + (size_t)(b * 128 + t) * 768;
            float rr = 1.0f / (1.0f + __expf(-(xrow[j]       + (float)gS[b * 784 + j]       + bhh[j])));
            float zz = 1.0f / (1.0f + __expf(-(xrow[256 + j] + (float)gS[b * 784 + 256 + j] + bhh[256 + j])));
            float nn = tanhf(xrow[512 + j] + rr * ((float)gS[b * 784 + 512 + j] + bhh[512 + j]));
            float hp = hS[b * 260 + j];
            float h2 = (1.0f - zz) * nn + zz * hp;
            hS[b * 260 + j]  = h2;
            h16[b * 264 + j] = (_Float16)h2;
            rm[idx] = fmaxf(rm[idx], h2);
        }
        __syncthreads();
    }
    for (int idx = tid; idx < 32 * 256; idx += 512) gout[idx] = rm[idx];
}

// ---------------------------------------------------------------------------
// Head: out[b] = sigmoid(W_label . |lvec - rvec| + b_label)
// ---------------------------------------------------------------------------
__global__ void finalK(const float* __restrict__ gmax, const float* __restrict__ Wlab,
                       const float* __restrict__ blab, float* __restrict__ out) {
    __shared__ float red[128];
    const int b = blockIdx.x, tid = threadIdx.x;
    float partial = 0.0f;
    for (int j = tid; j < 512; j += 128) {
        int dir = j >> 8, jj = j & 255;
        float lv = gmax[(size_t)((0 * 2 + dir) * 32 + b) * 256 + jj];
        float rv = gmax[(size_t)((1 * 2 + dir) * 32 + b) * 256 + jj];
        partial += Wlab[j] * fabsf(lv - rv);
    }
    red[tid] = partial;
    __syncthreads();
    for (int s = 64; s > 0; s >>= 1) {
        if (tid < s) red[tid] += red[tid + s];
        __syncthreads();
    }
    if (tid == 0) out[b] = 1.0f / (1.0f + expf(-(red[0] + blab[0])));
}

// ---------------------------------------------------------------------------
extern "C" void kernel_launch(void* const* d_in, const int* in_sizes, int n_in,
                              void* d_out, int out_size, void* d_ws, size_t ws_size,
                              hipStream_t stream) {
    (void)in_sizes; (void)n_in; (void)out_size; (void)ws_size;
    const int*   x1    = (const int*)d_in[0];
    const int*   x2    = (const int*)d_in[1];
    const float* emb   = (const float*)d_in[2];
    const float* W_c   = (const float*)d_in[3];
    const float* b_c   = (const float*)d_in[4];
    const float* W_ihf = (const float*)d_in[5];
    const float* W_hhf = (const float*)d_in[6];
    const float* b_ihf = (const float*)d_in[7];
    const float* b_hhf = (const float*)d_in[8];
    const float* W_ihb = (const float*)d_in[9];
    const float* W_hhb = (const float*)d_in[10];
    const float* b_ihb = (const float*)d_in[11];
    const float* b_hhb = (const float*)d_in[12];
    const float* Wlab  = (const float*)d_in[13];
    const float* blab  = (const float*)d_in[14];

    char* ws = (char*)d_ws;
    _Float16* Wc_h  = (_Float16*)(ws + 0);          //  16384 h
    _Float16* Wih_h = (_Float16*)(ws + 32768);      //  2 x 98304 h
    _Float16* Whh_h = (_Float16*)(ws + 425984);     //  2 x 196608 h
    _Float16* e_h   = (_Float16*)(ws + 1212416);    //  2 x 4096 x 128 h
    float*    xp    = (float*)   (ws + 3309568);    //  4 x 4096 x 768 f32
    float*    gmaxb = (float*)   (ws + 53641216);   //  4 x 32 x 256 f32

    cvtK<<<(16384  + 255) / 256, 256, 0, stream>>>(W_c,   Wc_h, 16384);
    cvtK<<<(98304  + 255) / 256, 256, 0, stream>>>(W_ihf, Wih_h,          98304);
    cvtK<<<(98304  + 255) / 256, 256, 0, stream>>>(W_ihb, Wih_h + 98304,  98304);
    cvtK<<<(196608 + 255) / 256, 256, 0, stream>>>(W_hhf, Whh_h,          196608);
    cvtK<<<(196608 + 255) / 256, 256, 0, stream>>>(W_hhb, Whh_h + 196608, 196608);

    treeK<<<8192, 256, 0, stream>>>(x1, x2, emb, Wc_h, b_c, e_h);
    xpK<<<dim3(64, 6, 4), 256, 0, stream>>>(e_h, Wih_h, b_ihf, b_ihb, xp);
    gruK<<<4, 512, 133120, stream>>>(xp, Whh_h, b_hhf, b_hhb, gmaxb);
    finalK<<<32, 128, 0, stream>>>(gmaxb, Wlab, blab, (float*)d_out);
}
